// LatentDynamicsModel_35321811042564
// MI455X (gfx1250) — compile-verified, matrix operand order checked by reading, not run on hardware
//
#include <hip/hip_runtime.h>
#include <hip/hip_bf16.h>

typedef __attribute__((ext_vector_type(16))) _Float16 v16h;
typedef __attribute__((ext_vector_type(8)))  _Float16 v8h;
typedef __attribute__((ext_vector_type(4)))  _Float16 v4h;
typedef __attribute__((ext_vector_type(8)))  float    v8f;

#define B_TOTAL 131072
#define LATENT  64
#define HID     256
#define ROWS    128          // batch rows per workgroup
#define BLOCK   128          // 4 waves; each wave owns 32 rows = two 16-row M-tiles
#define STR_X   104          // halves; stride of K=96 input buffer (16B-aligned, conflict-free)
#define STR_H   264          // halves; stride of H=256 buffers   (16B-aligned, conflict-free)

// workspace layout (halves): weights converted to f16 B-fragment layout
#define WC_IN   0
#define WC_IH   24576                          // W_in : Ks=3,  N=256 -> 3*16*512
#define WC_HH   (24576 + 196608)               // W_ih : Ks=8,  N=768 -> 8*48*512
#define WC_O1   (24576 + 2*196608)             // W_hh : same
#define WC_O2   (24576 + 2*196608 + 65536)     // W_o1 : Ks=8,  N=256
#define WC_TOTAL (24576 + 2*196608 + 65536 + 16384) // W_o2: Ks=8, N=64

// ---------------------------------------------------------------------------
// Weight conversion: src is [N][Ksrc] f32 row-major (PyTorch weight). We need
// B = W^T (K x N) in the CDNA5 16x16x32 f16 B-operand layout:
//   fragment f = nt*Ks + kt holds the 32x16 tile (k = kt*32.., n = nt*16..)
//   within a fragment: lane = (n%16) + 16*g, slot j: k = kt*32 + 8g + (j&7) + 16*(j>>3)
// ---------------------------------------------------------------------------
__global__ void convert_weights_kernel(const float* __restrict__ Win,
                                       const float* __restrict__ Wih,
                                       const float* __restrict__ Whh,
                                       const float* __restrict__ Wo1,
                                       const float* __restrict__ Wo2,
                                       _Float16* __restrict__ dst) {
  int idx = blockIdx.x * blockDim.x + threadIdx.x;
  if (idx >= WC_TOTAL) return;
  const float* src;
  int e, Ks, Ksrc;
  if (idx < WC_IH)      { src = Win; e = idx - WC_IN; Ks = 3; Ksrc = 66;  }
  else if (idx < WC_HH) { src = Wih; e = idx - WC_IH; Ks = 8; Ksrc = 256; }
  else if (idx < WC_O1) { src = Whh; e = idx - WC_HH; Ks = 8; Ksrc = 256; }
  else if (idx < WC_O2) { src = Wo1; e = idx - WC_O1; Ks = 8; Ksrc = 256; }
  else                  { src = Wo2; e = idx - WC_O2; Ks = 8; Ksrc = 256; }
  int frag = e >> 9, w = e & 511;
  int lane = w >> 4, j = w & 15;
  int nt = frag / Ks, kt = frag - nt * Ks;
  int n  = nt * 16 + (lane & 15);
  int g  = lane >> 4;
  int k  = kt * 32 + 8 * g + (j & 7) + 16 * (j >> 3);
  float v = (k < Ksrc) ? src[n * Ksrc + k] : 0.0f;
  dst[idx] = (_Float16)v;
}

// ---------------------------------------------------------------------------
// Fragment loaders
// A (16x32 f16): lane = m + 16*g, slots 0..7 -> k = 8g+0..7, slots 8..15 -> k = 8g+16..23
//   => two contiguous 16B chunks per lane from a row-major f16 LDS buffer.
// ---------------------------------------------------------------------------
__device__ __forceinline__ v16h load_a(const _Float16* buf, int stride, int kt) {
  int lane = threadIdx.x & 31;
  int m = lane & 15, g = lane >> 4;
  const _Float16* p = buf + m * stride + kt * 32 + 8 * g;
  v8h lo = *(const v8h*)p;
  v8h hi = *(const v8h*)(p + 16);
  return __builtin_shufflevector(lo, hi, 0,1,2,3,4,5,6,7,8,9,10,11,12,13,14,15);
}

__device__ __forceinline__ v16h load_b(const _Float16* w, int Ks, int nt, int kt) {
  int lane = threadIdx.x & 31;
  const _Float16* p = w + (((size_t)(nt * Ks + kt)) << 9) + (lane << 4);
  v8h lo = *(const v8h*)p;
  v8h hi = *(const v8h*)(p + 8);
  return __builtin_shufflevector(lo, hi, 0,1,2,3,4,5,6,7,8,9,10,11,12,13,14,15);
}

__device__ __forceinline__ v8f wmma(v16h a, v16h b, v8f c) {
  return __builtin_amdgcn_wmma_f32_16x16x32_f16(false, a, false, b, (short)0, c,
                                                false, false);
}

__device__ __forceinline__ v8f splat8(float s) {
  return (v8f){s, s, s, s, s, s, s, s};
}
__device__ __forceinline__ float fsig(float x) { return 1.0f / (1.0f + __expf(-x)); }
__device__ __forceinline__ float ftanh_(float x) {
  float e = __expf(-2.0f * x);
  return (1.0f - e) / (1.0f + e);
}

// ---------------------------------------------------------------------------
// Fused latent-dynamics kernel: one 128-row batch tile per workgroup.
// 4 waves; wave w owns rows 32w..32w+31 (two M-tiles) so every B fragment is
// loaded once and fed to two WMMAs (halves L2 weight traffic per FLOP).
// ---------------------------------------------------------------------------
extern "C" __global__ void __launch_bounds__(BLOCK)
ldm_fused(const float* __restrict__ z, const float* __restrict__ act,
          const float* __restrict__ hid,
          const float* __restrict__ b_in, const float* __restrict__ b_ih,
          const float* __restrict__ b_hh, const float* __restrict__ b_o1,
          const float* __restrict__ b_o2, const _Float16* __restrict__ wc,
          float* __restrict__ out_z, float* __restrict__ out_h) {
  __shared__ _Float16 smem[3 * ROWS * STR_H];          // ~198 KB
  _Float16* hbuf = smem;                               // h  (f16), stride STR_H
  _Float16* xbuf = smem + ROWS * STR_H;                // x, later y
  _Float16* nbuf = smem + 2 * ROWS * STR_H;            // xin (STR_X), later h_new

  const int tid  = threadIdx.x;
  const int lane = tid & 31;
  const int wave = tid >> 5;
  const int ln   = lane & 15;   // column within 16-wide n-tile (C/D lane -> N)
  const int lg   = lane >> 4;   // C/D m-half selector
  const size_t tile_row0 = (size_t)blockIdx.x * ROWS;

  // ---- cooperative fills: h -> hbuf f16 ; [z|action|0pad] -> nbuf (K=96) ----
  {
    const float4* hsrc = (const float4*)(hid + tile_row0 * HID);
    for (int i = tid; i < ROWS * HID / 4; i += BLOCK) {
      float4 v = hsrc[i];
      int row = i >> 6, c4 = i & 63;
      v4h p = {(_Float16)v.x, (_Float16)v.y, (_Float16)v.z, (_Float16)v.w};
      *(v4h*)(hbuf + row * STR_H + (c4 << 2)) = p;
    }
    const float4* zsrc = (const float4*)(z + tile_row0 * LATENT);
    for (int i = tid; i < ROWS * LATENT / 4; i += BLOCK) {
      float4 v = zsrc[i];
      int row = i >> 4, c4 = i & 15;
      v4h p = {(_Float16)v.x, (_Float16)v.y, (_Float16)v.z, (_Float16)v.w};
      *(v4h*)(nbuf + row * STR_X + (c4 << 2)) = p;
    }
    for (int i = tid; i < ROWS * 9; i += BLOCK) {     // zero pad cols 68..103
      int row = i / 9, c = 68 + (i % 9) * 4;
      v4h p = {};
      *(v4h*)(nbuf + row * STR_X + c) = p;
    }
    if (tid < ROWS) {                                 // action -> cols 64,65 (+zeros 66,67)
      const float2 a = ((const float2*)act)[tile_row0 + tid];
      v4h p = {(_Float16)a.x, (_Float16)a.y, (_Float16)0.0f, (_Float16)0.0f};
      *(v4h*)(nbuf + tid * STR_X + 64) = p;
    }
  }
  __syncthreads();

  const int wrow = wave * 32;                        // first row owned by wave
  const _Float16* axin = nbuf + wrow * STR_X;        // xin rows (two M-tiles)
  const _Float16* ahp  = hbuf + wrow * STR_H;        // h rows
  _Float16*       axp  = xbuf + wrow * STR_H;        // x / y rows
  _Float16*       anp  = nbuf + wrow * STR_H;        // h_new rows (overlays xin)

  // ---- stage 1: x = relu([z,a] @ W_in^T + b_in), K=96, N=256 ----
  for (int nt = 0; nt < 16; ++nt) {
    float bb = b_in[nt * 16 + ln];
    v8f a0 = splat8(bb), a1 = splat8(bb);
#pragma unroll
    for (int kt = 0; kt < 3; ++kt) {
      v16h b = load_b(wc + WC_IN, 3, nt, kt);
      a0 = wmma(load_a(axin, STR_X, kt), b, a0);
      a1 = wmma(load_a(axin + 16 * STR_X, STR_X, kt), b, a1);
    }
#pragma unroll
    for (int r = 0; r < 8; ++r) {
      axp[(r + 8 * lg) * STR_H + nt * 16 + ln]        = (_Float16)fmaxf(a0[r], 0.0f);
      axp[(16 + r + 8 * lg) * STR_H + nt * 16 + ln]   = (_Float16)fmaxf(a1[r], 0.0f);
    }
  }
  __syncthreads();   // all waves done reading xin before h_new overlays nbuf

  // ---- stage 2+3: GRU gates. gi/gh tiles live only in registers. ----
  const float* hrow  = hid   + (tile_row0 + wrow) * HID;
  float*       ohrow = out_h + (tile_row0 + wrow) * HID;
  for (int nt = 0; nt < 16; ++nt) {
    const int col = nt * 16 + ln;
    v8f ir0 = splat8(b_ih[col]),       ir1 = ir0;
    v8f iz0 = splat8(b_ih[col + 256]), iz1 = iz0;
    v8f in0 = splat8(b_ih[col + 512]), in1 = in0;
    v8f hr0 = splat8(b_hh[col]),       hr1 = hr0;
    v8f hz0 = splat8(b_hh[col + 256]), hz1 = hz0;
    v8f hn0 = splat8(b_hh[col + 512]), hn1 = hn0;
#pragma unroll
    for (int kt = 0; kt < 8; ++kt) {
      v16h ax0 = load_a(axp, STR_H, kt);
      v16h ax1 = load_a(axp + 16 * STR_H, STR_H, kt);
      v16h ah0 = load_a(ahp, STR_H, kt);
      v16h ah1 = load_a(ahp + 16 * STR_H, STR_H, kt);
      v16h b;
      b = load_b(wc + WC_IH, 8, nt,      kt); ir0 = wmma(ax0, b, ir0); ir1 = wmma(ax1, b, ir1);
      b = load_b(wc + WC_IH, 8, nt + 16, kt); iz0 = wmma(ax0, b, iz0); iz1 = wmma(ax1, b, iz1);
      b = load_b(wc + WC_IH, 8, nt + 32, kt); in0 = wmma(ax0, b, in0); in1 = wmma(ax1, b, in1);
      b = load_b(wc + WC_HH, 8, nt,      kt); hr0 = wmma(ah0, b, hr0); hr1 = wmma(ah1, b, hr1);
      b = load_b(wc + WC_HH, 8, nt + 16, kt); hz0 = wmma(ah0, b, hz0); hz1 = wmma(ah1, b, hz1);
      b = load_b(wc + WC_HH, 8, nt + 32, kt); hn0 = wmma(ah0, b, hn0); hn1 = wmma(ah1, b, hn1);
    }
    // epilogue for both M-tiles
#pragma unroll
    for (int mt = 0; mt < 2; ++mt) {
      const v8f& ir = mt ? ir1 : ir0;  const v8f& iz = mt ? iz1 : iz0;
      const v8f& inn = mt ? in1 : in0; const v8f& hr = mt ? hr1 : hr0;
      const v8f& hz = mt ? hz1 : hz0;  const v8f& hn = mt ? hn1 : hn0;
#pragma unroll
      for (int r = 0; r < 8; ++r) {
        int m = mt * 16 + r + 8 * lg;
        float rg = fsig(ir[r] + hr[r]);
        float zg = fsig(iz[r] + hz[r]);
        float ng = ftanh_(inn[r] + rg * hn[r]);
        float hv = hrow[m * HID + col];               // exact f32 h from L2
        float hnew = (1.0f - zg) * ng + zg * hv;
        anp[m * STR_H + col] = (_Float16)hnew;        // A-operand for stage 4
        ohrow[m * HID + col] = hnew;                  // output h_new (f32)
      }
    }
  }

  // ---- stage 4: y = relu(h_new @ W_o1^T + b_o1), K=256, N=256 ----
  for (int nt = 0; nt < 16; ++nt) {
    float bb = b_o1[nt * 16 + ln];
    v8f a0 = splat8(bb), a1 = splat8(bb);
#pragma unroll
    for (int kt = 0; kt < 8; ++kt) {
      v16h b = load_b(wc + WC_O1, 8, nt, kt);
      a0 = wmma(load_a(anp, STR_H, kt), b, a0);
      a1 = wmma(load_a(anp + 16 * STR_H, STR_H, kt), b, a1);
    }
#pragma unroll
    for (int r = 0; r < 8; ++r) {
      axp[(r + 8 * lg) * STR_H + nt * 16 + ln]      = (_Float16)fmaxf(a0[r], 0.0f);
      axp[(16 + r + 8 * lg) * STR_H + nt * 16 + ln] = (_Float16)fmaxf(a1[r], 0.0f);
    }
  }

  // ---- stage 5: z_next = y @ W_o2^T + b_o2, K=256, N=64 ----
  float* ozrow = out_z + (tile_row0 + wrow) * LATENT;
  for (int nt = 0; nt < 4; ++nt) {
    float bb = b_o2[nt * 16 + ln];
    v8f a0 = splat8(bb), a1 = splat8(bb);
#pragma unroll
    for (int kt = 0; kt < 8; ++kt) {
      v16h b = load_b(wc + WC_O2, 8, nt, kt);
      a0 = wmma(load_a(axp, STR_H, kt), b, a0);
      a1 = wmma(load_a(axp + 16 * STR_H, STR_H, kt), b, a1);
    }
#pragma unroll
    for (int r = 0; r < 8; ++r) {
      ozrow[(r + 8 * lg) * LATENT + nt * 16 + ln]        = a0[r];
      ozrow[(16 + r + 8 * lg) * LATENT + nt * 16 + ln]   = a1[r];
    }
  }
}

// ---------------------------------------------------------------------------
extern "C" void kernel_launch(void* const* d_in, const int* in_sizes, int n_in,
                              void* d_out, int out_size, void* d_ws, size_t ws_size,
                              hipStream_t stream) {
  const float* z    = (const float*)d_in[0];
  const float* act  = (const float*)d_in[1];
  const float* hid  = (const float*)d_in[2];
  const float* Win  = (const float*)d_in[3];
  const float* b_in = (const float*)d_in[4];
  const float* Wih  = (const float*)d_in[5];
  const float* Whh  = (const float*)d_in[6];
  const float* b_ih = (const float*)d_in[7];
  const float* b_hh = (const float*)d_in[8];
  const float* Wo1  = (const float*)d_in[9];
  const float* b_o1 = (const float*)d_in[10];
  const float* Wo2  = (const float*)d_in[11];
  const float* b_o2 = (const float*)d_in[12];

  _Float16* wc    = (_Float16*)d_ws;                  // ~1 MB of fragment weights
  float*    out_z = (float*)d_out;                    // [B,64]
  float*    out_h = out_z + (size_t)B_TOTAL * LATENT; // [B,256]

  convert_weights_kernel<<<(WC_TOTAL + 255) / 256, 256, 0, stream>>>(
      Win, Wih, Whh, Wo1, Wo2, wc);
  ldm_fused<<<B_TOTAL / ROWS, BLOCK, 0, stream>>>(
      z, act, hid, b_in, b_ih, b_hh, b_o1, b_o2, wc, out_z, out_h);
}
